// MultiHeadCrossAttention_39144331936377
// MI455X (gfx1250) — compile-verified, matrix operand order checked by reading, not run on hardware
//
#include <hip/hip_runtime.h>
#include <hip/hip_bf16.h>

// ---------------------------------------------------------------------------
// ProbSparse multi-head cross attention (Informer) for gfx1250 / MI455X.
//   B=4, L=4096, D_MODEL=512, H=8, D_HEAD=64, U_PART=N_TOP=45
// All four big GEMMs: bf16 WMMA (v_wmma_f32_16x16x32_bf16, f32 accumulate),
// operands pre-converted to bf16 once (weights also pre-transposed), tiles
// staged to LDS with async global->LDS copies, double-buffered.
// Sparse stages (sampled scores, top-k, 45-query softmax-attn): f32, L2-bound.
// ---------------------------------------------------------------------------

#define BDIM 4
#define LSEQ 4096
#define DMODEL 512
#define NHEADS 8
#define DHEAD 64
#define NTOP 45
#define UPART 45
#define MROWS (BDIM * LSEQ)          // 16384
#define LDSTRIDE 40                  // bf16 elems per LDS row (80B, bank-safe)

typedef __bf16 v16bf __attribute__((ext_vector_type(16)));
typedef __bf16 v8bf  __attribute__((ext_vector_type(8)));
typedef __bf16 bf2   __attribute__((ext_vector_type(2)));
typedef float  v8f   __attribute__((ext_vector_type(8)));
typedef int    v4i   __attribute__((ext_vector_type(4)));

// ---- async global->LDS copy (CDNA5), guarded so compilation never breaks ---
#if defined(__has_builtin)
#if __has_builtin(__builtin_amdgcn_global_load_async_to_lds_b128)
#define HAVE_ASYNC 1
#endif
#endif
#ifndef HAVE_ASYNC
#define HAVE_ASYNC 0
#endif

#if HAVE_ASYNC
// builtin expects: (v4i AS(1)* gaddr, v4i AS(3)* ldsaddr, imm offset, imm cpol)
#define ASYNC_CP16(gp, lp)                                                     \
  __builtin_amdgcn_global_load_async_to_lds_b128(                              \
      (__attribute__((address_space(1))) v4i*)(gp),                            \
      (__attribute__((address_space(3))) v4i*)(lp), 0, 0)
#if __has_builtin(__builtin_amdgcn_s_wait_asynccnt)
#define WAIT_ASYNC(n) __builtin_amdgcn_s_wait_asynccnt(n)
#else
#define WAIT_ASYNC(n) asm volatile("s_wait_asynccnt %0" ::"i"(n) : "memory")
#endif
#else
#define ASYNC_CP16(gp, lp) (*(float4*)(void*)(lp) = *(const float4*)(const void*)(gp))
#define WAIT_ASYNC(n) ((void)0)
#endif

// ---------------------------------------------------------------------------
// f32 -> bf16 elementwise (8 elems/thread, 16B stores)
// ---------------------------------------------------------------------------
__global__ __launch_bounds__(256)
void cvt_bf16_kernel(const float* __restrict__ in, __bf16* __restrict__ out)
{
  size_t i = ((size_t)blockIdx.x * 256 + threadIdx.x) * 8;
  float4 a = *(const float4*)(in + i);
  float4 b = *(const float4*)(in + i + 4);
  v8bf o;
  o[0] = (__bf16)a.x; o[1] = (__bf16)a.y; o[2] = (__bf16)a.z; o[3] = (__bf16)a.w;
  o[4] = (__bf16)b.x; o[5] = (__bf16)b.y; o[6] = (__bf16)b.z; o[7] = (__bf16)b.w;
  *(v8bf*)(out + i) = o;
}

// ---------------------------------------------------------------------------
// W[512,512] f32 -> WT[n][k] bf16 (transpose so B-fragments read contiguous K)
// ---------------------------------------------------------------------------
__global__ __launch_bounds__(256)
void transpose_bf16_kernel(const float* __restrict__ W, __bf16* __restrict__ WT)
{
  __shared__ float t[32][33];
  const int tx = threadIdx.x, ty = threadIdx.y;     // (32, 8)
  const int bn = blockIdx.x * 32;                    // n tile
  const int bk = blockIdx.y * 32;                    // k tile
  #pragma unroll
  for (int i = 0; i < 4; ++i)
    t[ty + i * 8][tx] = W[(size_t)(bk + ty + i * 8) * DMODEL + bn + tx];
  __syncthreads();
  #pragma unroll
  for (int i = 0; i < 4; ++i)
    WT[(size_t)(bn + ty + i * 8) * DMODEL + bk + tx] = (__bf16)t[tx][ty + i * 8];
}

// ---------------------------------------------------------------------------
// GEMM: C[16384,512] = A[16384,512](bf16) * WT'[512,512](bf16, [n][k]) + bias
//   AMODE 0: A row-major [m*512+k];  1: A is bf16 ctx in (B,H,L,64) layout
//   OMODE 0: f32 head-split out [((b*8+h)*4096+l)*64+d];  1: flat [m*512+n]
// 256 thr (8 waves). WG tile 128x64, wave 32x32 = 2x2 WMMA. K step 32,
// double-buffered async LDS staging.
// ---------------------------------------------------------------------------
template <int AMODE, int OMODE>
__global__ __launch_bounds__(256)
void gemm512_bf16(const __bf16* __restrict__ A, const __bf16* __restrict__ WT,
                  const float* __restrict__ bias, float* __restrict__ out)
{
  __shared__ __bf16 sA[2][128][LDSTRIDE];   // 2 x 128x32 tile
  __shared__ __bf16 sW[2][64][LDSTRIDE];    // 2 x 64x32 tile ([n][k])

  const int tid  = threadIdx.x;
  const int lane = tid & 31;
  const int wid  = tid >> 5;
  const int wm   = wid & 3;
  const int wn   = wid >> 2;
  const int blockM = blockIdx.y * 128;
  const int blockN = blockIdx.x * 64;

  // per-thread copy slots: A chunks 512 (2/thread), W chunks 256 (1/thread)
  const int ar0 = tid >> 2,         ac0 = (tid & 3) << 3;          // rows 0..63
  const int ar1 = (tid + 256) >> 2, ac1 = ((tid + 256) & 3) << 3;  // rows 64..127
  const int wr  = tid >> 2,         wc  = (tid & 3) << 3;          // n 0..63

  auto a_addr = [&](int row, int k) -> const __bf16* {
    int gm = blockM + row;
    if (AMODE == 0) return A + (size_t)gm * DMODEL + k;
    int b = gm >> 12, l = gm & 4095, h = k >> 6, d = k & 63;
    return A + ((size_t)((b << 3) + h) * LSEQ + l) * DHEAD + d;
  };
  auto issue = [&](int buf, int k0) {
    ASYNC_CP16(a_addr(ar0, k0 + ac0), &sA[buf][ar0][ac0]);
    ASYNC_CP16(a_addr(ar1, k0 + ac1), &sA[buf][ar1][ac1]);
    ASYNC_CP16(WT + (size_t)(blockN + wr) * DMODEL + k0 + wc, &sW[buf][wr][wc]);
  };

  v8f acc[2][2] = {};
  const int sel = lane & 15;
  const int kh  = (lane >> 4) << 3;   // 0 or 8

  issue(0, 0);
  for (int s = 0; s < DMODEL / 32; ++s) {
    const int cur = s & 1;
    const bool more = (s + 1) < (DMODEL / 32);
    if (more) issue(cur ^ 1, (s + 1) * 32);
    if (more) { WAIT_ASYNC(3); } else { WAIT_ASYNC(0); }
    __syncthreads();

    v16bf af[2], bfv[2];
    #pragma unroll
    for (int t = 0; t < 2; ++t) {
      const int r = wm * 32 + t * 16 + sel;
      const int c = wn * 32 + t * 16 + sel;
      #pragma unroll
      for (int j = 0; j < 8; ++j) {
        int kk = ((j & 4) ? 16 : 0) + kh + ((j & 3) << 1);
        bf2 pa = *(const bf2*)&sA[cur][r][kk];
        af[t][2 * j] = pa.x;  af[t][2 * j + 1] = pa.y;
        bf2 pb = *(const bf2*)&sW[cur][c][kk];
        bfv[t][2 * j] = pb.x; bfv[t][2 * j + 1] = pb.y;
      }
    }
    #pragma unroll
    for (int tm = 0; tm < 2; ++tm)
      #pragma unroll
      for (int tn = 0; tn < 2; ++tn)
        acc[tm][tn] = __builtin_amdgcn_wmma_f32_16x16x32_bf16(
            false, af[tm], false, bfv[tn], (short)0, acc[tm][tn], false, false);
    __syncthreads();
  }

  const int nsel = lane & 15;
  const int mh   = (lane >> 4) << 3;
  #pragma unroll
  for (int tm = 0; tm < 2; ++tm) {
    #pragma unroll
    for (int tn = 0; tn < 2; ++tn) {
      int gn = blockN + wn * 32 + tn * 16 + nsel;
      float bv = bias[gn];
      #pragma unroll
      for (int r = 0; r < 8; ++r) {
        int gm = blockM + wm * 32 + tm * 16 + mh + r;
        float v = acc[tm][tn][r] + bv;
        if (OMODE == 0) {
          int b = gm >> 12, l = gm & 4095, h = gn >> 6, d = gn & 63;
          out[((size_t)((b << 3) + h) * LSEQ + l) * DHEAD + d] = v;
        } else {
          out[(size_t)gm * DMODEL + gn] = v;
        }
      }
    }
  }
}

// ---------------------------------------------------------------------------
// V column means per (b,h)
// ---------------------------------------------------------------------------
__global__ __launch_bounds__(256)
void vmean_kernel(const float* __restrict__ V, float* __restrict__ vmean)
{
  __shared__ float red[256];
  const int bh = blockIdx.x, tid = threadIdx.x;
  const int d = tid & 63, slice = tid >> 6;
  const float* Vb = V + (size_t)bh * LSEQ * DHEAD;
  float s = 0.f;
  for (int k = slice; k < LSEQ; k += 4) s += Vb[(size_t)k * DHEAD + d];
  red[tid] = s;
  __syncthreads();
  if (tid < 64)
    vmean[bh * DHEAD + tid] =
        (red[tid] + red[tid + 64] + red[tid + 128] + red[tid + 192]) *
        (1.0f / (float)LSEQ);
}

// ctx[b,h,l,d] = bf16(vmean[b,h,d]) — 8 elems/thread
__global__ __launch_bounds__(256)
void ctx_init_kernel(__bf16* __restrict__ ctx, const float* __restrict__ vmean)
{
  size_t i8 = ((size_t)blockIdx.x * 256 + threadIdx.x) * 8;
  int d  = (int)(i8 & 63);
  int bh = (int)(i8 >> 18);
  v8bf o;
  #pragma unroll
  for (int j = 0; j < 8; ++j) o[j] = (__bf16)vmean[bh * DHEAD + d + j];
  *(v8bf*)(ctx + i8) = o;
}

// ---------------------------------------------------------------------------
// Sampled sparsity scores: one wave per (b,h,l) row.
// ---------------------------------------------------------------------------
__global__ __launch_bounds__(256)
void sampled_scores_kernel(const float* __restrict__ Q, const float* __restrict__ K,
                           const int* __restrict__ sidx, float* __restrict__ Mout)
{
  const int lane = threadIdx.x & 31;
  const int wid  = threadIdx.x >> 5;
  const int gid  = blockIdx.x * 8 + wid;
  const int bh = gid >> 12, l = gid & 4095;

  const float* qrow = Q + (size_t)gid * DHEAD;
  const float q0 = qrow[lane];
  const float q1 = qrow[lane + 32];
  const float* Kb = K + (size_t)bh * LSEQ * DHEAD;
  const int* su = sidx + (size_t)l * UPART;

  float mx = -3.4e38f, sm = 0.f;
  for (int u = 0; u < UPART; ++u) {
    const float* kr = Kb + (size_t)su[u] * DHEAD;
    float p = q0 * kr[lane] + q1 * kr[lane + 32];
    #pragma unroll
    for (int off = 16; off > 0; off >>= 1) p += __shfl_xor(p, off, 32);
    mx = fmaxf(mx, p);
    sm += p;
  }
  if (lane == 0) Mout[gid] = mx - sm * (1.0f / (float)LSEQ);
}

// ---------------------------------------------------------------------------
// Top-45 per (b,h), iterative argmax (ties -> lowest index)
// ---------------------------------------------------------------------------
__global__ __launch_bounds__(256)
void topk_kernel(const float* __restrict__ Mbuf, int* __restrict__ topi)
{
  __shared__ float vals[LSEQ];
  __shared__ float rv[256];
  __shared__ int   ri[256];
  const int bh = blockIdx.x, tid = threadIdx.x;
  for (int i = tid; i < LSEQ; i += 256) vals[i] = Mbuf[(size_t)bh * LSEQ + i];
  __syncthreads();

  for (int t = 0; t < NTOP; ++t) {
    float bv = -3.4e38f;
    int bi = 0x7fffffff;
    for (int i = tid; i < LSEQ; i += 256) {
      float v = vals[i];
      if (v > bv || (v == bv && i < bi)) { bv = v; bi = i; }
    }
    rv[tid] = bv; ri[tid] = bi;
    __syncthreads();
    for (int s = 128; s > 0; s >>= 1) {
      if (tid < s) {
        if (rv[tid + s] > rv[tid] ||
            (rv[tid + s] == rv[tid] && ri[tid + s] < ri[tid])) {
          rv[tid] = rv[tid + s]; ri[tid] = ri[tid + s];
        }
      }
      __syncthreads();
    }
    if (tid == 0) {
      topi[bh * NTOP + t] = ri[0];
      vals[ri[0]] = -3.4e38f;
    }
    __syncthreads();
  }
}

// ---------------------------------------------------------------------------
// Dense attention for 45 selected queries; scatter bf16 row into ctx.
// ---------------------------------------------------------------------------
__global__ __launch_bounds__(256)
void sparse_attn_kernel(const float* __restrict__ Q, const float* __restrict__ K,
                        const float* __restrict__ V, const int* __restrict__ topi,
                        __bf16* __restrict__ ctx)
{
  __shared__ float qs[DHEAD];
  __shared__ float sc[LSEQ];
  __shared__ float red[256];

  const int blk = blockIdx.x;
  const int bh = blk / NTOP, u = blk % NTOP;
  const int tid = threadIdx.x;
  const int l = topi[bh * NTOP + u];

  const float* Kb = K + (size_t)bh * LSEQ * DHEAD;
  const float* Vb = V + (size_t)bh * LSEQ * DHEAD;
  if (tid < DHEAD) qs[tid] = Q[((size_t)bh * LSEQ + l) * DHEAD + tid];
  __syncthreads();

  float lmax = -3.4e38f;
  for (int k = tid; k < LSEQ; k += 256) {
    const float* kr = Kb + (size_t)k * DHEAD;
    float s = 0.f;
    #pragma unroll
    for (int d = 0; d < DHEAD; ++d) s += qs[d] * kr[d];
    s *= 0.125f;
    sc[k] = s;
    lmax = fmaxf(lmax, s);
  }
  red[tid] = lmax; __syncthreads();
  for (int s = 128; s > 0; s >>= 1) {
    if (tid < s) red[tid] = fmaxf(red[tid], red[tid + s]);
    __syncthreads();
  }
  const float gmax = red[0];
  __syncthreads();

  float lsum = 0.f;
  for (int k = tid; k < LSEQ; k += 256) {
    float e = __expf(sc[k] - gmax);
    sc[k] = e;
    lsum += e;
  }
  red[tid] = lsum; __syncthreads();
  for (int s = 128; s > 0; s >>= 1) {
    if (tid < s) red[tid] += red[tid + s];
    __syncthreads();
  }
  const float inv = 1.0f / red[0];
  __syncthreads();

  const int d = tid & 63, slice = tid >> 6;
  float a = 0.f;
  for (int k = slice; k < LSEQ; k += 4) a += sc[k] * Vb[(size_t)k * DHEAD + d];
  red[tid] = a; __syncthreads();
  if (tid < DHEAD) {
    float v = (red[tid] + red[tid + 64] + red[tid + 128] + red[tid + 192]) * inv;
    ctx[((size_t)bh * LSEQ + l) * DHEAD + tid] = (__bf16)v;
  }
}

// ---------------------------------------------------------------------------
// Host launch
// ---------------------------------------------------------------------------
extern "C" void kernel_launch(void* const* d_in, const int* in_sizes, int n_in,
                              void* d_out, int out_size, void* d_ws, size_t ws_size,
                              hipStream_t stream)
{
  const float* x    = (const float*)d_in[0];
  const float* cin  = (const float*)d_in[1];
  const float* Wq   = (const float*)d_in[2];
  const float* bq   = (const float*)d_in[3];
  const float* Wk   = (const float*)d_in[4];
  const float* bk   = (const float*)d_in[5];
  const float* Wv   = (const float*)d_in[6];
  const float* bv   = (const float*)d_in[7];
  const float* Wo   = (const float*)d_in[8];
  const float* bo   = (const float*)d_in[9];
  const int*   sidx = (const int*)d_in[10];

  char* ws = (char*)d_ws;
  const size_t SZ_BF   = (size_t)MROWS * DMODEL * sizeof(__bf16);   // 16.78 MB
  const size_t SZ_WT   = (size_t)DMODEL * DMODEL * sizeof(__bf16);  // 0.52 MB
  const size_t SZ_F32  = (size_t)MROWS * DMODEL * sizeof(float);    // 33.55 MB

  size_t off = 0;
  __bf16* xbf  = (__bf16*)(ws + off); off += SZ_BF;
  __bf16* cbf  = (__bf16*)(ws + off); off += SZ_BF;
  __bf16* WTq  = (__bf16*)(ws + off); off += SZ_WT;
  __bf16* WTk  = (__bf16*)(ws + off); off += SZ_WT;
  __bf16* WTv  = (__bf16*)(ws + off); off += SZ_WT;
  __bf16* WTo  = (__bf16*)(ws + off); off += SZ_WT;
  float*  Q    = (float*)(ws + off);  off += SZ_F32;
  float*  K    = (float*)(ws + off);  off += SZ_F32;
  float*  V    = (float*)(ws + off);  off += SZ_F32;
  __bf16* CTX  = (__bf16*)(ws + off); off += SZ_BF;
  float*  Mbuf = (float*)(ws + off);  off += (size_t)BDIM * NHEADS * LSEQ * 4;
  float*  vme  = (float*)(ws + off);  off += (size_t)BDIM * NHEADS * DHEAD * 4;
  int*    topi = (int*)(ws + off);

  dim3 gg(DMODEL / 64, MROWS / 128);   // (8, 128)
  dim3 bb(256);
  dim3 tg(16, 16), tb(32, 8);

  cvt_bf16_kernel<<<(MROWS * DMODEL) / 2048, 256, 0, stream>>>(x, xbf);
  cvt_bf16_kernel<<<(MROWS * DMODEL) / 2048, 256, 0, stream>>>(cin, cbf);
  transpose_bf16_kernel<<<tg, tb, 0, stream>>>(Wq, WTq);
  transpose_bf16_kernel<<<tg, tb, 0, stream>>>(Wk, WTk);
  transpose_bf16_kernel<<<tg, tb, 0, stream>>>(Wv, WTv);
  transpose_bf16_kernel<<<tg, tb, 0, stream>>>(Wo, WTo);

  gemm512_bf16<0, 0><<<gg, bb, 0, stream>>>(xbf, WTq, bq, Q);
  gemm512_bf16<0, 0><<<gg, bb, 0, stream>>>(cbf, WTk, bk, K);
  gemm512_bf16<0, 0><<<gg, bb, 0, stream>>>(cbf, WTv, bv, V);

  vmean_kernel<<<BDIM * NHEADS, 256, 0, stream>>>(V, vme);
  ctx_init_kernel<<<(MROWS * DMODEL) / 2048, 256, 0, stream>>>(CTX, vme);

  sampled_scores_kernel<<<(BDIM * NHEADS * LSEQ) / 8, 256, 0, stream>>>(Q, K, sidx, Mbuf);
  topk_kernel<<<BDIM * NHEADS, 256, 0, stream>>>(Mbuf, topi);
  sparse_attn_kernel<<<BDIM * NHEADS * NTOP, 256, 0, stream>>>(Q, K, V, topi, CTX);

  gemm512_bf16<1, 1><<<gg, bb, 0, stream>>>(CTX, WTo, bo, (float*)d_out);
}
